// FusedLinearCrossEntropyLoss_8796093022765
// MI455X (gfx1250) — matrix operand order repaired and checked by reading, not compiled, not run on hardware
//
#include <hip/hip_runtime.h>
#include <math.h>

// Fused lm_head projection + cross-entropy (mean over valid rows).
// logits = hidden(4096x2048 f32) @ weight^T(32000x2048 f32); online LSE, no logits in HBM.
// Fast path: one-shot f32->bf16 convert into d_ws, then WMMA GEMM whose LDS staging is
// done by the CDNA5 Tensor Data Mover (tensor_load_to_lds, TENSORcnt), double-buffered.

typedef __attribute__((ext_vector_type(16))) __bf16 v16bf;
typedef __attribute__((ext_vector_type(8)))  float  v8f;
typedef __attribute__((ext_vector_type(4)))  unsigned int v4u;
typedef __attribute__((ext_vector_type(8)))  unsigned int v8u;

constexpr int BT   = 4096;
constexpr int Hdim = 2048;
constexpr int Vdim = 32000;
constexpr int IGNORE_INDEX = -100;

constexpr int ROWS = 64;           // BT rows per block
constexpr int COLS = 256;          // vocab cols per block
constexpr int KB   = 64;           // K per LDS stage
constexpr int NCH  = Vdim / COLS;  // 125 vocab chunks

constexpr int LDA = 72;            // padded LDS row stride (bf16 elems): 128B row + 16B pad
constexpr int LDB = 72;
constexpr int LDC = 257;           // logits LDS row stride (f32)

constexpr int A_BYTES = ROWS * LDA * 2;    // 9216
constexpr int B_BYTES = COLS * LDB * 2;    // 36864
constexpr int L_BYTES = ROWS * LDC * 4;    // 65792
constexpr int SMEM_F32  = L_BYTES;                 // staging(46080) overlaid by logits
constexpr int SMEM_BF16 = 2 * (A_BYTES + B_BYTES); // 92160 double-buffered; logits overlay

static __device__ __forceinline__ unsigned short f2bf(float f) {
  unsigned int u = __float_as_uint(f);
  unsigned int r = u + 0x7FFFu + ((u >> 16) & 1u);   // round-to-nearest-even
  return (unsigned short)(r >> 16);
}

union FragU { uint4 q[2]; v16bf v; };

// ---- shared compute helpers -------------------------------------------------

static __device__ __forceinline__ void mma_ksteps(const unsigned short* sA,
                                                  const unsigned short* sB,
                                                  int wrow, int wcol, int l16, int half,
                                                  v8f acc[2][4]) {
#pragma unroll
  for (int ks = 0; ks < KB; ks += 32) {
    FragU afr[2];
#pragma unroll
    for (int i = 0; i < 2; ++i) {
      // A 16x32 bf16 layout: half 0 -> K 0-7,16-23 ; half 1 -> K 8-15,24-31
      int base = (wrow + i * 16 + l16) * LDA + ks + 8 * half;
      afr[i].q[0] = *(const uint4*)(sA + base);
      afr[i].q[1] = *(const uint4*)(sA + base + 16);
    }
    FragU bfr[4];
#pragma unroll
    for (int j = 0; j < 4; ++j) {
      // B 32x16 bf16 layout: half 0 -> K 0-7,16-23 ; half 1 -> K 8-15,24-31
      int base = (wcol + j * 16 + l16) * LDB + ks + 8 * half;
      bfr[j].q[0] = *(const uint4*)(sB + base);
      bfr[j].q[1] = *(const uint4*)(sB + base + 16);
    }
#pragma unroll
    for (int i = 0; i < 2; ++i)
#pragma unroll
      for (int j = 0; j < 4; ++j)
        acc[i][j] = __builtin_amdgcn_wmma_f32_16x16x32_bf16(
            false, afr[i].v, false, bfr[j].v, (short)0, acc[i][j], false, false);
  }
}

static __device__ __forceinline__ void lse_epilogue(float* sL, v8f acc[2][4],
                                                    int tid, int lane, int wrow, int wcol,
                                                    int l16, int half, int rowBase, int vBase,
                                                    int chunk, const int* __restrict__ targets,
                                                    float* __restrict__ part_m,
                                                    float* __restrict__ part_s,
                                                    float* __restrict__ tgt_logit) {
  __syncthreads();
#pragma unroll
  for (int i = 0; i < 2; ++i)
#pragma unroll
    for (int j = 0; j < 4; ++j)
#pragma unroll
      for (int r = 0; r < 8; ++r) {
        int m = wrow + i * 16 + r + 8 * half;    // C layout: VGPR r, lane half
        int n = wcol + j * 16 + l16;
        sL[m * LDC + n] = acc[i][j][r];
      }
  __syncthreads();

  const int rrow = tid >> 2;         // 4 threads per row, 64 cols each
  const int seg  = tid & 3;
  const float* lr = sL + rrow * LDC + seg * 64;

  float m = -__builtin_inff();
#pragma unroll 8
  for (int i = 0; i < 64; ++i) {
    int c = (i + seg * 16) & 63;     // rotated scan -> conflict-free LDS banks
    m = fmaxf(m, lr[c]);
  }
  m = fmaxf(m, __shfl_xor(m, 1, 32));
  m = fmaxf(m, __shfl_xor(m, 2, 32));

  float s = 0.0f;
#pragma unroll 8
  for (int i = 0; i < 64; ++i) {
    int c = (i + seg * 16) & 63;
    s += expf(lr[c] - m);
  }
  s += __shfl_xor(s, 1, 32);
  s += __shfl_xor(s, 2, 32);

  if (seg == 0) {
    int rowG = rowBase + rrow;
    part_m[(size_t)chunk * BT + rowG] = m;
    part_s[(size_t)chunk * BT + rowG] = s;
    int tg = targets[rowG];
    if (tg >= vBase && tg < vBase + COLS)     // exactly one chunk matches per row
      tgt_logit[rowG] = sL[rrow * LDC + (tg - vBase)];
  }
}

// ---- TDM tile load: one instruction moves an entire 2D bf16 tile to LDS -----
// D# per CDNA5 ISA ch.8: group0 = {count, lds_addr, global_addr, type=2},
// group1 = {data_size=2B, pad(interval=32 dwords, amount=4 dwords -> LDS stride 144B),
//           tensor_dim0/1, tile_dim0/1, tensor_dim0_stride}. Groups 2/3 zero (2D tile).
static __device__ __forceinline__ void tdm_tile_load(unsigned lds_byte_addr,
                                                     const unsigned short* gptr,
                                                     unsigned tensor_d0, unsigned tensor_d1,
                                                     unsigned tile_d0, unsigned tile_d1,
                                                     unsigned stride0) {
  unsigned long long ga = (unsigned long long)(uintptr_t)gptr;
  v4u g0;
  g0.x = 1u;                                   // count=1 valid descriptor
  g0.y = lds_byte_addr;
  g0.z = (unsigned)ga;
  g0.w = (unsigned)(ga >> 32) | (2u << 30);    // type=2 ("image")
  v8u g1;
  g1[0] = (1u << 16)                           // data_size = 2 bytes
        | (1u << 20)                           // pad_enable
        | (4u << 22)                           // pad_interval: every 32 DWORDs (128B row)
        | (3u << 25);                          // pad_amount: 4 DWORDs (16B)
  g1[1] = (tensor_d0 & 0xFFFFu) << 16;                          // tensor_dim0 [79:48]
  g1[2] = (tensor_d0 >> 16) | ((tensor_d1 & 0xFFFFu) << 16);    // tensor_dim1 [111:80]
  g1[3] = (tensor_d1 >> 16) | (tile_d0 << 16);                  // tile_dim0  [127:112]
  g1[4] = tile_d1;                                              // tile_dim1, tile_dim2=0
  g1[5] = stride0;                                              // dim0_stride lo
  g1[6] = 0u;
  g1[7] = 0u;
  v4u gz = {0u, 0u, 0u, 0u};
  asm volatile("tensor_load_to_lds %0, %1, %2, %3"
               :: "s"(g0), "s"(g1), "s"(gz), "s"(gz)
               : "memory");
}

// ---- fast path: bf16 operands staged by the Tensor Data Mover ---------------

__global__ __launch_bounds__(256)
void flce_gemm_bf16(const unsigned short* __restrict__ bfA,   // 4096x2048 bf16
                    const unsigned short* __restrict__ bfW,   // 32000x2048 bf16
                    const int* __restrict__ targets,
                    float* __restrict__ part_m, float* __restrict__ part_s,
                    float* __restrict__ tgt_logit) {
  __shared__ __align__(16) unsigned char smem[SMEM_BF16];
  float* sL = (float*)smem;

  const int tid  = threadIdx.x;
  const int lane = tid & 31;
  const int wv   = tid >> 5;
  const int wrow = (wv >> 2) * 32;
  const int wcol = (wv & 3) * 64;
  const int half = lane >> 4;
  const int l16  = lane & 15;

  const int rowBase = blockIdx.x * ROWS;   // x fastest => weight chunk reused via L2
  const int vBase   = blockIdx.y * COLS;

  v8f acc[2][4];
#pragma unroll
  for (int i = 0; i < 2; ++i)
#pragma unroll
    for (int j = 0; j < 4; ++j)
#pragma unroll
      for (int r = 0; r < 8; ++r) acc[i][j][r] = 0.0f;

  const unsigned ldsBase = (unsigned)(uintptr_t)(void*)smem;   // LDS byte address of smem
  const int S = Hdim / KB;                 // 32 stages
  auto issue = [&](int s) {
    unsigned boff = (s & 1) ? (unsigned)(A_BYTES + B_BYTES) : 0u;   // ping-pong
    tdm_tile_load(ldsBase + boff,
                  bfA + (size_t)rowBase * Hdim + s * KB,
                  Hdim, BT, KB, ROWS, Hdim);
    tdm_tile_load(ldsBase + boff + A_BYTES,
                  bfW + (size_t)vBase * Hdim + s * KB,
                  Hdim, Vdim, KB, COLS, Hdim);
  };

  if (wv == 0) issue(0);
  for (int s = 0; s < S; ++s) {
    if (wv == 0) {
      if (s + 1 < S) {
        issue(s + 1);                      // prefetch next stage into other buffer
        __builtin_amdgcn_s_wait_tensorcnt(2);  // oldest 2 (stage s) complete
      } else {
        __builtin_amdgcn_s_wait_tensorcnt(0);  // drain
      }
    }
    __syncthreads();                       // release all waves: stage s data visible
    unsigned boff = (s & 1) ? (unsigned)(A_BYTES + B_BYTES) : 0u;
    const unsigned short* sA = (const unsigned short*)(smem + boff);
    const unsigned short* sB = (const unsigned short*)(smem + boff + A_BYTES);
    mma_ksteps(sA, sB, wrow, wcol, l16, half, acc);
    __syncthreads();                       // everyone done before this buffer is reused
  }

  lse_epilogue(sL, acc, tid, lane, wrow, wcol, l16, half, rowBase, vBase,
               blockIdx.y, targets, part_m, part_s, tgt_logit);
}

__global__ __launch_bounds__(256)
void convert_bf16(const float* __restrict__ src, unsigned short* __restrict__ dst, int n4) {
  int i = blockIdx.x * blockDim.x + threadIdx.x;
  int stride = gridDim.x * blockDim.x;
  for (; i < n4; i += stride) {
    float4 f = ((const float4*)src)[i];
    union { unsigned short u[4]; uint2 v; } pk;
    pk.u[0] = f2bf(f.x); pk.u[1] = f2bf(f.y); pk.u[2] = f2bf(f.z); pk.u[3] = f2bf(f.w);
    ((uint2*)dst)[i] = pk.v;
  }
}

// ---- fallback path: f32 global loads + in-loop conversion (round-1 kernel) --

__global__ __launch_bounds__(256)
void flce_gemm_f32(const float* __restrict__ hs, const float* __restrict__ wgt,
                   const int* __restrict__ targets,
                   float* __restrict__ part_m, float* __restrict__ part_s,
                   float* __restrict__ tgt_logit) {
  __shared__ __align__(16) unsigned char smem[SMEM_F32];
  unsigned short* sA = (unsigned short*)smem;
  unsigned short* sB = (unsigned short*)(smem + A_BYTES);
  float* sL = (float*)smem;

  const int tid  = threadIdx.x;
  const int lane = tid & 31;
  const int wv   = tid >> 5;
  const int wrow = (wv >> 2) * 32;
  const int wcol = (wv & 3) * 64;
  const int half = lane >> 4;
  const int l16  = lane & 15;

  const int rowBase = blockIdx.x * ROWS;
  const int vBase   = blockIdx.y * COLS;

  v8f acc[2][4];
#pragma unroll
  for (int i = 0; i < 2; ++i)
#pragma unroll
    for (int j = 0; j < 4; ++j)
#pragma unroll
      for (int r = 0; r < 8; ++r) acc[i][j][r] = 0.0f;

  for (int kk = 0; kk < Hdim; kk += KB) {
    __syncthreads();
#pragma unroll
    for (int i = 0; i < 4; ++i) {
      int j  = tid + 256 * i;
      int r  = j >> 4;
      int c4 = j & 15;
      float4 f = *(const float4*)(hs + (size_t)(rowBase + r) * Hdim + kk + c4 * 4);
      union { unsigned short u[4]; uint2 v; } pk;
      pk.u[0] = f2bf(f.x); pk.u[1] = f2bf(f.y); pk.u[2] = f2bf(f.z); pk.u[3] = f2bf(f.w);
      *(uint2*)(sA + r * LDA + c4 * 4) = pk.v;
    }
#pragma unroll
    for (int i = 0; i < 16; ++i) {
      int j  = tid + 256 * i;
      int r  = j >> 4;
      int c4 = j & 15;
      float4 f = *(const float4*)(wgt + (size_t)(vBase + r) * Hdim + kk + c4 * 4);
      union { unsigned short u[4]; uint2 v; } pk;
      pk.u[0] = f2bf(f.x); pk.u[1] = f2bf(f.y); pk.u[2] = f2bf(f.z); pk.u[3] = f2bf(f.w);
      *(uint2*)(sB + r * LDB + c4 * 4) = pk.v;
    }
    __syncthreads();
    mma_ksteps(sA, sB, wrow, wcol, l16, half, acc);
  }

  lse_epilogue(sL, acc, tid, lane, wrow, wcol, l16, half, rowBase, vBase,
               blockIdx.y, targets, part_m, part_s, tgt_logit);
}

// ---- combine / reduce kernels ----------------------------------------------

__global__ __launch_bounds__(256)
void flce_rows(const float* __restrict__ part_m, const float* __restrict__ part_s,
               const float* __restrict__ tgt_logit, const int* __restrict__ targets,
               float* __restrict__ rloss, float* __restrict__ rvalid) {
  int row = blockIdx.x * blockDim.x + threadIdx.x;
  if (row >= BT) return;
  float M = -__builtin_inff();
  for (int c = 0; c < NCH; ++c) M = fmaxf(M, part_m[(size_t)c * BT + row]);
  float S = 0.0f;
  for (int c = 0; c < NCH; ++c)
    S += part_s[(size_t)c * BT + row] * expf(part_m[(size_t)c * BT + row] - M);
  float lse = M + logf(S);
  int tg = targets[row];
  bool valid = (tg != IGNORE_INDEX);
  rloss[row]  = valid ? (lse - tgt_logit[row]) : 0.0f;
  rvalid[row] = valid ? 1.0f : 0.0f;
}

__global__ __launch_bounds__(1024)
void flce_final(const float* __restrict__ rloss, const float* __restrict__ rvalid,
                float* __restrict__ out) {
  __shared__ float sl[1024];
  __shared__ float sv[1024];
  int t = threadIdx.x;
  float a = 0.0f, b = 0.0f;
  for (int i = t; i < BT; i += 1024) { a += rloss[i]; b += rvalid[i]; }
  sl[t] = a; sv[t] = b;
  __syncthreads();
  for (int off = 512; off > 0; off >>= 1) {
    if (t < off) { sl[t] += sl[t + off]; sv[t] += sv[t + off]; }
    __syncthreads();
  }
  if (t == 0) out[0] = sl[0] / fmaxf(sv[0], 1.0f);
}

// ---- host-side orchestration -----------------------------------------------

extern "C" void kernel_launch(void* const* d_in, const int* in_sizes, int n_in,
                              void* d_out, int out_size, void* d_ws, size_t ws_size,
                              hipStream_t stream) {
  (void)in_sizes; (void)n_in; (void)out_size;
  const float* hs  = (const float*)d_in[0];
  const float* wgt = (const float*)d_in[1];
  const int*   tgt = (const int*)d_in[2];
  float* out = (float*)d_out;

  float* wsf    = (float*)d_ws;
  float* part_m = wsf;                            // NCH*BT
  float* part_s = wsf + (size_t)NCH * BT;         // NCH*BT
  float* tgt_l  = wsf + 2 * (size_t)NCH * BT;     // BT
  float* rloss  = tgt_l + BT;                     // BT
  float* rvalid = rloss + BT;                     // BT
  size_t part_bytes = (2 * (size_t)NCH * BT + 3 * BT) * sizeof(float);

  size_t bfA_off   = (part_bytes + 255) & ~(size_t)255;
  size_t bfA_bytes = (size_t)BT * Hdim * 2;       // 16 MB
  size_t bfW_off   = bfA_off + bfA_bytes;
  size_t bfW_bytes = (size_t)Vdim * Hdim * 2;     // 128 MB
  size_t need      = bfW_off + bfW_bytes;

  if (ws_size >= need) {
    unsigned short* bfA = (unsigned short*)((char*)d_ws + bfA_off);
    unsigned short* bfW = (unsigned short*)((char*)d_ws + bfW_off);
    convert_bf16<<<4096, 256, 0, stream>>>(hs, bfA, BT * Hdim / 4);
    convert_bf16<<<8192, 256, 0, stream>>>(wgt, bfW, Vdim * Hdim / 4);
    flce_gemm_bf16<<<dim3(BT / ROWS, NCH), 256, 0, stream>>>(bfA, bfW, tgt,
                                                             part_m, part_s, tgt_l);
  } else {
    flce_gemm_f32<<<dim3(BT / ROWS, NCH), 256, 0, stream>>>(hs, wgt, tgt,
                                                            part_m, part_s, tgt_l);
  }
  flce_rows<<<dim3((BT + 255) / 256), 256, 0, stream>>>(part_m, part_s, tgt_l, tgt,
                                                        rloss, rvalid);
  flce_final<<<1, 1024, 0, stream>>>(rloss, rvalid, out);
}
